// VQVAE_10222022165029
// MI455X (gfx1250) — matrix operand order repaired and checked by reading, not compiled, not run on hardware
//
#include <hip/hip_runtime.h>
#include <hip/hip_bf16.h>

typedef __attribute__((ext_vector_type(16))) _Float16 v16h;
typedef __attribute__((ext_vector_type(8)))  float    v8f;
typedef __attribute__((ext_vector_type(2)))  int      v2i;
typedef __attribute__((ext_vector_type(4)))  int      v4i;

#define GLOBAL_AS __attribute__((address_space(1)))
#define LDS_AS    __attribute__((address_space(3)))

// gfx1250 async global->LDS copy path (ASYNCcnt), guarded so either toolchain compiles.
#if defined(__has_builtin)
#if __has_builtin(__builtin_amdgcn_global_load_async_to_lds_b64) && \
    __has_builtin(__builtin_amdgcn_global_load_async_to_lds_b128) && \
    __has_builtin(__builtin_amdgcn_s_wait_asynccnt)
#define HAVE_ASYNC_LDS 1
#endif
#endif
#ifndef HAVE_ASYNC_LDS
#define HAVE_ASYNC_LDS 0
#endif

__device__ __forceinline__ v8f wmma16x16x32(v16h a, v16h b, v8f c) {
  return __builtin_amdgcn_wmma_f32_16x16x32_f16(
      false, a, false, b, (short)0, c, false, false);
}

// 16-bit A 16x32 fragment: lane l row m=l&15; element h -> K = 16*(l>>4) + h (contiguous)
__device__ __forceinline__ int kmap_a(int h, int lane) {
  return ((lane >> 4) << 4) | h;
}

#if HAVE_ASYNC_LDS
__device__ __forceinline__ void async_copy_b64(const _Float16* g, _Float16* l) {
  __builtin_amdgcn_global_load_async_to_lds_b64(
      (GLOBAL_AS v2i*)g, (LDS_AS v2i*)l, 0, 0);
}
__device__ __forceinline__ void async_copy_b128(const _Float16* g, _Float16* l) {
  __builtin_amdgcn_global_load_async_to_lds_b128(
      (GLOBAL_AS v4i*)g, (LDS_AS v4i*)l, 0, 0);
}
#endif

// ===========================================================================
// Weight packing: f32 (Cout,Cin,kh,kw) -> f16 B16[CoutPad][Kpad], K-contig,
// zero padded. (co,ci,ky,kx) is contiguous, so src index = co*Ktot + K.
// ===========================================================================
__global__ __launch_bounds__(256) void pack_w_kernel(
    const float* __restrict__ w, _Float16* __restrict__ dst,
    int Cout, int Ktot, int Kpad, int CoutPad)
{
  const int t = blockIdx.x * 256 + threadIdx.x;
  if (t >= CoutPad * Kpad) return;
  const int co = t / Kpad, K = t - co * Kpad;
  float v = (co < Cout && K < Ktot) ? w[co * Ktot + K] : 0.f;
  dst[t] = (_Float16)v;
}

// Transposed-conv weights (Cin,Cout,4,4) -> f16 Bt[par][CoutPad][Cin*4],
// one B matrix per output parity class (py,px); tap set fixed per class.
__global__ __launch_bounds__(256) void pack_wt_kernel(
    const float* __restrict__ w, _Float16* __restrict__ dst,
    int Cin, int Cout, int CoutPad)
{
  const int Ktot = Cin << 2;
  const int t = blockIdx.x * 256 + threadIdx.x;
  if (t >= 4 * CoutPad * Ktot) return;
  const int par = t / (CoutPad * Ktot);
  int r = t - par * (CoutPad * Ktot);
  const int co = r / Ktot, K = r - co * Ktot;
  const int py = par & 1, px = par >> 1;
  const int ty = (py + 1) & 1, tx = (px + 1) & 1;
  const int ci = K >> 2, tap = K & 3;
  const int ky = ty + ((tap & 1) << 1), kx = tx + ((tap >> 1) << 1);
  float v = (co < Cout) ? w[((ci * Cout + co) << 4) + (ky << 2) + kx] : 0.f;
  dst[t] = (_Float16)v;
}

__global__ __launch_bounds__(256) void pack_f16_kernel(
    const float* __restrict__ src, _Float16* __restrict__ dst, int n)
{
  const int t = blockIdx.x * 256 + threadIdx.x;
  if (t < n) dst[t] = (_Float16)src[t];
}

// ===========================================================================
// Direct conv, implicit GEMM. Per wave: one 16-pixel M tile x NSUB 16-cout
// tiles (one A gather feeds NSUB WMMAs). Weights async-copied to LDS.
// flags: 1=ReLU on input read, 2=ReLU out, 4=add residual
// ===========================================================================
template <int KH, int KW, int NSUB>
__global__ __launch_bounds__(128) void conv_wmma_kernel(
    const float* __restrict__ in, const _Float16* __restrict__ w16,
    const float* __restrict__ bias, const float* __restrict__ res,
    float* __restrict__ out,
    int N, int Cin, int H, int W, int Cout, int Ho, int Wo,
    int stride, int pad, int Kpad, int flags)
{
  __shared__ alignas(32) _Float16 sB[NSUB * 16 * 32];
  const int lane = threadIdx.x & 31;
  const int wave = threadIdx.x >> 5;
  const int n0 = blockIdx.y * (NSUB * 16);
  const int m0 = ((blockIdx.x << 2) + wave) << 4;
  constexpr int KHKW = KH * KW;
  const int Ktot = Cin * KHKW;
  const int HoWo = Ho * Wo;

  const int mrow = m0 + (lane & 15);
  int pn, poy, pox;
  { int t = mrow; pn = t / HoWo; t -= pn * HoWo; poy = t / Wo; pox = t - poy * Wo; }
  const int oyb = poy * stride - pad;
  const int oxb = pox * stride - pad;
  const float* inP = in + (size_t)pn * Cin * H * W;
  const bool relu_in = (flags & 1) != 0;

  v8f acc[NSUB] = {};
  for (int kk = 0; kk < Kpad; kk += 32) {
    __syncthreads();
#if HAVE_ASYNC_LDS
    // 64B per B16 row: NSUB*16 rows x 8 b64 segments, one async op per lane
    for (int seg = threadIdx.x; seg < NSUB * 16 * 8; seg += 128) {
      const int row = seg >> 3, s4 = (seg & 7) << 2;
      async_copy_b64(w16 + (size_t)(n0 + row) * Kpad + kk + s4,
                     sB + row * 32 + s4);
    }
    __builtin_amdgcn_s_wait_asynccnt(0);
#else
    for (int e = threadIdx.x; e < NSUB * 16 * 32; e += 128) {
      const int row = e >> 5, kloc = e & 31;
      sB[e] = w16[(size_t)(n0 + row) * Kpad + kk + kloc];
    }
#endif
    __syncthreads();

    // A fragment: per-lane im2col gather (constant-divisor decode)
    v16h a;
#pragma unroll
    for (int h = 0; h < 16; ++h) {
      const int K = kk + kmap_a(h, lane);
      float av = 0.f;
      if (K < Ktot) {
        const int ci = K / KHKW;
        const int rr = K - ci * KHKW;
        const int ky = rr / KW, kx = rr - ky * KW;
        const int iy = oyb + ky, ix = oxb + kx;
        if ((unsigned)iy < (unsigned)H && (unsigned)ix < (unsigned)W) {
          av = inP[(ci * H + iy) * W + ix];
          if (relu_in) av = fmaxf(av, 0.f);
        }
      }
      a[h] = (_Float16)av;
    }
#pragma unroll
    for (int s = 0; s < NSUB; ++s) {
      const v16h b =
          *(const v16h*)(sB + ((s << 4) + (lane & 15)) * 32 + ((lane >> 4) << 4));
      acc[s] = wmma16x16x32(a, b, acc[s]);
    }
  }

  // Store: col co = n0+s*16+(lane&15); rows m = m0 + r + 8*(lane>>4)
  int p0[8];
#pragma unroll
  for (int r = 0; r < 8; ++r) {
    const int m = m0 + r + ((lane >> 4) << 3);
    int t = m; const int n = t / HoWo; t -= n * HoWo;   // t = oy*Wo+ox
    p0[r] = n * Cout * HoWo + t;
  }
#pragma unroll
  for (int s = 0; s < NSUB; ++s) {
    const int co = n0 + (s << 4) + (lane & 15);
    if (co < Cout) {
      const float bv = bias[co];
#pragma unroll
      for (int r = 0; r < 8; ++r) {
        float v = acc[s][r] + bv;
        const int off = p0[r] + co * HoWo;
        if (flags & 4) v += res[off];
        if (flags & 2) v = fmaxf(v, 0.f);
        out[off] = v;
      }
    }
  }
}

// ===========================================================================
// Transposed conv k4 s2 p1, parity-partitioned gather GEMM; packed per-parity
// weights Bt[par][CoutPad][Cin*4]. grid.z = parity class. flags: 2=ReLU out.
// ===========================================================================
template <int NSUB>
__global__ __launch_bounds__(128) void convt_wmma_kernel(
    const float* __restrict__ in, const _Float16* __restrict__ w16,
    const float* __restrict__ bias, float* __restrict__ out,
    int N, int Cin, int H, int W, int Cout, int Ho, int Wo,
    int CoutPad, int flags)
{
  __shared__ alignas(32) _Float16 sB[NSUB * 16 * 32];
  const int lane = threadIdx.x & 31;
  const int wave = threadIdx.x >> 5;
  const int py = blockIdx.z & 1, px = blockIdx.z >> 1;
  const int n0 = blockIdx.y * (NSUB * 16);
  const int Hh = Ho >> 1, Wh = Wo >> 1;
  const int m0 = ((blockIdx.x << 2) + wave) << 4;
  const int Ktot = Cin << 2;
  const _Float16* wPar = w16 + (size_t)blockIdx.z * CoutPad * Ktot;

  const int mrow = m0 + (lane & 15);
  int pn, oy, ox;
  { int t = mrow; pn = t / (Hh * Wh); t -= pn * Hh * Wh;
    int yh = t / Wh; int xh = t - yh * Wh; oy = (yh << 1) + py; ox = (xh << 1) + px; }
  const int ty = (oy + 1) & 1, tx = (ox + 1) & 1;
  const float* inP = in + (size_t)pn * Cin * H * W;

  v8f acc[NSUB] = {};
  for (int kk = 0; kk < Ktot; kk += 32) {
    __syncthreads();
#if HAVE_ASYNC_LDS
    for (int seg = threadIdx.x; seg < NSUB * 16 * 8; seg += 128) {
      const int row = seg >> 3, s4 = (seg & 7) << 2;
      async_copy_b64(wPar + (size_t)(n0 + row) * Ktot + kk + s4,
                     sB + row * 32 + s4);
    }
    __builtin_amdgcn_s_wait_asynccnt(0);
#else
    for (int e = threadIdx.x; e < NSUB * 16 * 32; e += 128) {
      const int row = e >> 5, kloc = e & 31;
      sB[e] = wPar[(size_t)(n0 + row) * Ktot + kk + kloc];
    }
#endif
    __syncthreads();

    v16h a;
#pragma unroll
    for (int h = 0; h < 16; ++h) {
      const int K = kk + kmap_a(h, lane);
      const int ci = K >> 2, tap = K & 3;
      const int ky = ty + ((tap & 1) << 1), kx = tx + ((tap >> 1) << 1);
      const int iy = (oy + 1 - ky) >> 1, ix = (ox + 1 - kx) >> 1;
      float av = 0.f;
      if ((unsigned)iy < (unsigned)H && (unsigned)ix < (unsigned)W)
        av = inP[(ci * H + iy) * W + ix];
      a[h] = (_Float16)av;
    }
#pragma unroll
    for (int s = 0; s < NSUB; ++s) {
      const v16h b =
          *(const v16h*)(sB + ((s << 4) + (lane & 15)) * 32 + ((lane >> 4) << 4));
      acc[s] = wmma16x16x32(a, b, acc[s]);
    }
  }

  int p0[8];
#pragma unroll
  for (int r = 0; r < 8; ++r) {
    const int m = m0 + r + ((lane >> 4) << 3);
    int t = m; const int n = t / (Hh * Wh); t -= n * Hh * Wh;
    const int yh = t / Wh, xh = t - yh * Wh;
    p0[r] = n * Cout * Ho * Wo + ((yh << 1) + py) * Wo + ((xh << 1) + px);
  }
#pragma unroll
  for (int s = 0; s < NSUB; ++s) {
    const int co = n0 + (s << 4) + (lane & 15);
    if (co < Cout) {
      const float bv = bias[co];
#pragma unroll
      for (int r = 0; r < 8; ++r) {
        float v = acc[s][r] + bv;
        if (flags & 2) v = fmaxf(v, 0.f);
        out[p0[r] + co * Ho * Wo] = v;
      }
    }
  }
}

// ===========================================================================
// VQ argmin: argmin_k (||c_k||^2 - 2 z.c_k). f16 codebook async-staged to LDS
// in two 32KB passes; 2 WMMAs per 16-code tile; shfl_xor row-argmin.
// ===========================================================================
__global__ __launch_bounds__(128) void vq_argmin_kernel(
    const float* __restrict__ z, const _Float16* __restrict__ cb16,
    int* __restrict__ idx)
{
  __shared__ alignas(32) _Float16 scb[256 * 64];
  __shared__ float snorm[256];
  const int lane = threadIdx.x & 31;
  const int wave = threadIdx.x >> 5;
  const int m0 = ((blockIdx.x << 2) + wave) << 4;
  const int mrow = m0 + (lane & 15);
  const int tn = mrow >> 12, ty = (mrow >> 6) & 63, tx = mrow & 63;
  const int sp = 64 * 64;

  v16h a0, a1;
#pragma unroll
  for (int h = 0; h < 16; ++h) {
    const int d = kmap_a(h, lane);
    a0[h] = (_Float16)z[(tn * 64 + d) * sp + ty * 64 + tx];
    a1[h] = (_Float16)z[(tn * 64 + d + 32) * sp + ty * 64 + tx];
  }

  float bestd[8]; int besti[8];
#pragma unroll
  for (int r = 0; r < 8; ++r) { bestd[r] = 3.4e38f; besti[r] = 0; }

  for (int pass = 0; pass < 2; ++pass) {
    __syncthreads();
#if HAVE_ASYNC_LDS
    for (int seg = threadIdx.x; seg < 2048; seg += 128)   // 16B per op
      async_copy_b128(cb16 + (pass << 14) + (seg << 3), scb + (seg << 3));
    __builtin_amdgcn_s_wait_asynccnt(0);
#else
    for (int e = threadIdx.x; e < 256 * 64; e += 128)
      scb[e] = cb16[(pass << 14) + e];
#endif
    __syncthreads();
    for (int k = threadIdx.x; k < 256; k += 128) {
      float s = 0.f;
      for (int d = 0; d < 64; ++d) { float v = (float)scb[k * 64 + d]; s += v * v; }
      snorm[k] = s;
    }
    __syncthreads();

    for (int j = 0; j < 16; ++j) {
      const int codeLoc = (j << 4) + (lane & 15);
      const v16h b0 = *(const v16h*)(scb + codeLoc * 64 + ((lane >> 4) << 4));
      const v16h b1 = *(const v16h*)(scb + codeLoc * 64 + 32 + ((lane >> 4) << 4));
      v8f dot = {};
      dot = wmma16x16x32(a0, b0, dot);
      dot = wmma16x16x32(a1, b1, dot);
      const float cn = snorm[codeLoc];
      const int codeGlob = (pass << 8) + codeLoc;
#pragma unroll
      for (int r = 0; r < 8; ++r) {
        const float dst = cn - 2.f * dot[r];
        if (dst < bestd[r]) { bestd[r] = dst; besti[r] = codeGlob; }
      }
    }
  }

#pragma unroll
  for (int r = 0; r < 8; ++r) {
    float d = bestd[r]; int i = besti[r];
    for (int s = 1; s < 16; s <<= 1) {
      const float od = __shfl_xor(d, s, 32);
      const int   oi = __shfl_xor(i, s, 32);
      if (od < d || (od == d && oi < i)) { d = od; i = oi; }
    }
    if ((lane & 15) == 0)
      idx[m0 + r + ((lane >> 4) << 3)] = i;
  }
}

__global__ __launch_bounds__(256) void vq_gather_loss_kernel(
    const float* __restrict__ z, const float* __restrict__ cb,
    const int* __restrict__ idx, float* __restrict__ q,
    float* __restrict__ lossAcc)
{
  const int t = blockIdx.x * 256 + threadIdx.x;
  const int m = t >> 6, d = t & 63;
  const int tn = m >> 12, yy = (m >> 6) & 63, xx = m & 63;
  const float qv = cb[idx[m] * 64 + d];
  const int zo = (tn * 64 + d) * 4096 + yy * 64 + xx;
  const float diff = qv - z[zo];
  q[zo] = qv;
  float v = diff * diff;
  for (int s = 16; s; s >>= 1) v += __shfl_xor(v, s, 32);
  if ((threadIdx.x & 31) == 0) atomicAdd(lossAcc, v);
}

__global__ void zero_loss_kernel(float* __restrict__ lossAcc) { lossAcc[0] = 0.f; }
__global__ void loss_final_kernel(const float* __restrict__ lossAcc,
                                  float* __restrict__ out) {
  out[0] = lossAcc[0] * (1.25f / (65536.f * 64.f));
}

// ===========================================================================
template <int KH, int KW, int NSUB>
static void launch_conv(const float* in, const _Float16* w16, const float* b,
                        const float* res, float* out, int N, int Cin, int H,
                        int W, int Cout, int Ho, int Wo, int stride, int pad,
                        int Kpad, int flags, hipStream_t stream) {
  dim3 g((unsigned)(N * Ho * Wo / 64), (unsigned)((Cout + NSUB * 16 - 1) / (NSUB * 16)));
  conv_wmma_kernel<KH, KW, NSUB><<<g, 128, 0, stream>>>(
      in, w16, b, res, out, N, Cin, H, W, Cout, Ho, Wo, stride, pad, Kpad, flags);
}
template <int NSUB>
static void launch_convt(const float* in, const _Float16* w16, const float* b,
                         float* out, int N, int Cin, int H, int W, int Cout,
                         int Ho, int Wo, int CoutPad, int flags, hipStream_t stream) {
  dim3 g((unsigned)(N * (Ho / 2) * (Wo / 2) / 64),
         (unsigned)(CoutPad / (NSUB * 16)), 4);
  convt_wmma_kernel<NSUB><<<g, 128, 0, stream>>>(
      in, w16, b, out, N, Cin, H, W, Cout, Ho, Wo, CoutPad, flags);
}

extern "C" void kernel_launch(void* const* d_in, const int* in_sizes, int n_in,
                              void* d_out, int out_size, void* d_ws, size_t ws_size,
                              hipStream_t stream) {
  (void)in_sizes; (void)n_in; (void)out_size; (void)ws_size;
  const float* x       = (const float*)d_in[0];
  const float* enc_w1  = (const float*)d_in[1];
  const float* enc_b1  = (const float*)d_in[2];
  const float* enc_w2  = (const float*)d_in[3];
  const float* enc_b2  = (const float*)d_in[4];
  const float* enc_w3  = (const float*)d_in[5];
  const float* enc_b3  = (const float*)d_in[6];
  const float* enc_rw1 = (const float*)d_in[7];
  const float* enc_rb1 = (const float*)d_in[8];
  const float* enc_rw2 = (const float*)d_in[9];
  const float* enc_rb2 = (const float*)d_in[10];
  const float* pre_w   = (const float*)d_in[11];
  const float* pre_b   = (const float*)d_in[12];
  const float* cbk     = (const float*)d_in[13];
  const float* dec_w1  = (const float*)d_in[14];
  const float* dec_b1  = (const float*)d_in[15];
  const float* dec_rw1 = (const float*)d_in[16];
  const float* dec_rb1 = (const float*)d_in[17];
  const float* dec_rw2 = (const float*)d_in[18];
  const float* dec_rb2 = (const float*)d_in[19];
  const float* dec_tw1 = (const float*)d_in[20];
  const float* dec_tb1 = (const float*)d_in[21];
  const float* dec_tw2 = (const float*)d_in[22];
  const float* dec_tb2 = (const float*)d_in[23];

  float* loss  = (float*)d_out;
  float* recon = (float*)d_out + 1;

  // ---- Workspace layout ----
  char* wsb = (char*)d_ws;
  size_t off = 0;
  auto allocH = [&](size_t n) { _Float16* p = (_Float16*)(wsb + off); off += n * 2; return p; };
  // Packed f16 weights (all sizes multiples of 64 halves -> 128B aligned)
  _Float16* p_enc1  = allocH(64 * 64);          // Kpad 64 (Ktot 48)
  _Float16* p_enc2  = allocH(128 * 1024);
  _Float16* p_enc3  = allocH(128 * 1152);
  _Float16* p_erc3a = allocH(32 * 1152);
  _Float16* p_erc1a = allocH(128 * 32);
  _Float16* p_erc3b = allocH(32 * 1152);
  _Float16* p_erc1b = allocH(128 * 32);
  _Float16* p_pre   = allocH(64 * 128);
  _Float16* p_dec1  = allocH(128 * 576);
  _Float16* p_drc3a = allocH(32 * 1152);
  _Float16* p_drc1a = allocH(128 * 32);
  _Float16* p_drc3b = allocH(32 * 1152);
  _Float16* p_drc1b = allocH(128 * 32);
  _Float16* p_dt1   = allocH(4 * 64 * 512);
  _Float16* p_dt2   = allocH(4 * 16 * 256);
  _Float16* p_cb    = allocH(512 * 64);
  off = (off + 255) & ~(size_t)255;
  float* bufA = (float*)(wsb + off); off += (size_t)16777216 * 4;  // 16*64*128*128
  float* bufB = (float*)(wsb + off); off += (size_t)8388608 * 4;   // 16*128*64*64
  float* bufC = (float*)(wsb + off); off += (size_t)8388608 * 4;
  float* bufD = (float*)(wsb + off); off += (size_t)2097152 * 4;   // 16*32*64*64
  float* bufZ = (float*)(wsb + off); off += (size_t)4194304 * 4;   // 16*64*64*64
  float* bufQ = (float*)(wsb + off); off += (size_t)4194304 * 4;
  int*   vidx = (int*)(wsb + off);   off += (size_t)65536 * 4;
  float* lacc = (float*)(wsb + off); off += 4;

  auto packw = [&](const float* w, _Float16* dst, int Cout, int Ktot, int Kpad, int CoutPad) {
    int tot = CoutPad * Kpad;
    pack_w_kernel<<<dim3((tot + 255) / 256), 256, 0, stream>>>(w, dst, Cout, Ktot, Kpad, CoutPad);
  };

  // ---- Pack weights to f16 GEMM layout ----
  packw(enc_w1, p_enc1, 64, 48, 64, 64);
  packw(enc_w2, p_enc2, 128, 1024, 1024, 128);
  packw(enc_w3, p_enc3, 128, 1152, 1152, 128);
  packw(enc_rw1,              p_erc3a, 32, 1152, 1152, 32);
  packw(enc_rw1 + 32*128*9,   p_erc3b, 32, 1152, 1152, 32);
  packw(enc_rw2,              p_erc1a, 128, 32, 32, 128);
  packw(enc_rw2 + 128*32,     p_erc1b, 128, 32, 32, 128);
  packw(pre_w, p_pre, 64, 128, 128, 64);
  packw(dec_w1, p_dec1, 128, 576, 576, 128);
  packw(dec_rw1,              p_drc3a, 32, 1152, 1152, 32);
  packw(dec_rw1 + 32*128*9,   p_drc3b, 32, 1152, 1152, 32);
  packw(dec_rw2,              p_drc1a, 128, 32, 32, 128);
  packw(dec_rw2 + 128*32,     p_drc1b, 128, 32, 32, 128);
  pack_wt_kernel<<<dim3((4*64*512 + 255) / 256), 256, 0, stream>>>(dec_tw1, p_dt1, 128, 64, 64);
  pack_wt_kernel<<<dim3((4*16*256 + 255) / 256), 256, 0, stream>>>(dec_tw2, p_dt2, 64, 3, 16);
  pack_f16_kernel<<<dim3(128), 256, 0, stream>>>(cbk, p_cb, 32768);

  // ---- Encoder ----
  launch_conv<4,4,4>(x,    p_enc1, enc_b1, nullptr, bufA, 16,   3, 256, 256,  64, 128, 128, 2, 1,   64, 2, stream);
  launch_conv<4,4,4>(bufA, p_enc2, enc_b2, nullptr, bufB, 16,  64, 128, 128, 128,  64,  64, 2, 1, 1024, 2, stream);
  launch_conv<3,3,4>(bufB, p_enc3, enc_b3, nullptr, bufC, 16, 128,  64,  64, 128,  64,  64, 1, 1, 1152, 0, stream);
  launch_conv<3,3,2>(bufC, p_erc3a, enc_rb1,       nullptr, bufD, 16, 128, 64, 64,  32, 64, 64, 1, 1, 1152, 1, stream);
  launch_conv<1,1,4>(bufD, p_erc1a, enc_rb2,       bufC,    bufB, 16,  32, 64, 64, 128, 64, 64, 1, 0,   32, 5, stream);
  launch_conv<3,3,2>(bufB, p_erc3b, enc_rb1 + 32,  nullptr, bufD, 16, 128, 64, 64,  32, 64, 64, 1, 1, 1152, 1, stream);
  launch_conv<1,1,4>(bufD, p_erc1b, enc_rb2 + 128, bufB,    bufC, 16,  32, 64, 64, 128, 64, 64, 1, 0,   32, 7, stream);
  launch_conv<1,1,4>(bufC, p_pre, pre_b, nullptr, bufZ, 16, 128, 64, 64, 64, 64, 64, 1, 0, 128, 0, stream);

  // ---- Vector quantization ----
  vq_argmin_kernel<<<dim3(1024), 128, 0, stream>>>(bufZ, p_cb, vidx);
  zero_loss_kernel<<<dim3(1), 1, 0, stream>>>(lacc);
  vq_gather_loss_kernel<<<dim3(16384), 256, 0, stream>>>(bufZ, cbk, vidx, bufQ, lacc);
  loss_final_kernel<<<dim3(1), 1, 0, stream>>>(lacc, loss);

  // ---- Decoder ----
  launch_conv<3,3,4>(bufQ, p_dec1, dec_b1, nullptr, bufB, 16, 64, 64, 64, 128, 64, 64, 1, 1, 576, 0, stream);
  launch_conv<3,3,2>(bufB, p_drc3a, dec_rb1,       nullptr, bufD, 16, 128, 64, 64,  32, 64, 64, 1, 1, 1152, 1, stream);
  launch_conv<1,1,4>(bufD, p_drc1a, dec_rb2,       bufB,    bufC, 16,  32, 64, 64, 128, 64, 64, 1, 0,   32, 5, stream);
  launch_conv<3,3,2>(bufC, p_drc3b, dec_rb1 + 32,  nullptr, bufD, 16, 128, 64, 64,  32, 64, 64, 1, 1, 1152, 1, stream);
  launch_conv<1,1,4>(bufD, p_drc1b, dec_rb2 + 128, bufC,    bufB, 16,  32, 64, 64, 128, 64, 64, 1, 0,   32, 7, stream);
  launch_convt<4>(bufB, p_dt1, dec_tb1, bufA,  16, 128,  64,  64, 64, 128, 128, 64, 2, stream);
  launch_convt<1>(bufA, p_dt2, dec_tb2, recon, 16,  64, 128, 128,  3, 256, 256, 16, 0, stream);
}
